// MultiHeadedAttention_29076928594267
// MI455X (gfx1250) — matrix operand order repaired and checked by reading, not compiled, not run on hardware
//
#include <hip/hip_runtime.h>

// MI455X / gfx1250, wave32. Fused LoRA-MHA pipeline:
//  1) fold LoRA into dense weights (pre-transposed, f16)
//  2) convert activations f32->f16
//  3) WMMA GEMMs (async-to-LDS double-buffered): Q/K/V proj -> scores ->
//     column softmax -> PV -> out proj
typedef __attribute__((ext_vector_type(16))) _Float16 v16h;
typedef __attribute__((ext_vector_type(8)))  float    v8f;
typedef int v4i __attribute__((vector_size(16)));

#define BB 4
#define SS 1024
#define DD 1024
#define HH 16
#define HDD 128
#define RR 16
#define PP 2048
#define MTOT (BB*SS)        // 4096

#define AS1 __attribute__((address_space(1)))
#define AS3 __attribute__((address_space(3)))

#if defined(__gfx1250__) && __has_builtin(__builtin_amdgcn_global_load_async_to_lds_b128)
#define USE_ASYNC_LDS 1
#else
#define USE_ASYNC_LDS 0
#endif

__device__ __forceinline__ void wait_async0() {
#if __has_builtin(__builtin_amdgcn_s_wait_asynccnt)
  __builtin_amdgcn_s_wait_asynccnt(0);
#else
  asm volatile("s_wait_asynccnt 0x0" ::: "memory");
#endif
}

union Frag { v16h h; uint4 q[2]; };

// ---------------- elementwise helpers ----------------
__global__ void cvt_f32_to_f16_x4(const float4* __restrict__ in,
                                  _Float16* __restrict__ out, int n4) {
  int i = blockIdx.x * blockDim.x + threadIdx.x;
  if (i >= n4) return;
  float4 v = in[i];
  _Float16 h[4] = {(_Float16)v.x, (_Float16)v.y, (_Float16)v.z, (_Float16)v.w};
  *(uint2*)(out + (size_t)i * 4) = *(uint2*)h;
}

// WT[n*K + k] = W[k*N + n] + sum_r A[k*R + r] * Bl[r*N + n]   (transposed f16 weight)
__global__ void fuse_weight_T(const float* __restrict__ W, const float* __restrict__ A,
                              const float* __restrict__ Bl, _Float16* __restrict__ WT,
                              int K, int N) {
  int idx = blockIdx.x * blockDim.x + threadIdx.x;
  if (idx >= K * N) return;
  int k = idx / N, n = idx - k * N;       // coalesced reads over n
  float v = W[(size_t)k * N + n];
  if (A) {
#pragma unroll
    for (int r = 0; r < RR; ++r) v += A[k * RR + r] * Bl[(size_t)r * N + n];
  }
  WT[(size_t)n * K + k] = (_Float16)v;
}

// softmax over the QUERY axis (column-wise): p[:,c] = softmax(s[:,c]) per head z
__global__ __launch_bounds__(256) void col_softmax(_Float16* __restrict__ Pm) {
  size_t z = blockIdx.y;
  int c = blockIdx.x * 256 + threadIdx.x;
  _Float16* p = Pm + z * (size_t)(SS * SS) + c;
  float m = -3.0e38f, Z = 0.f;
  for (int r = 0; r < SS; ++r) {
    float s = (float)p[(size_t)r * SS];
    float nm = fmaxf(m, s);
    Z = Z * __expf(m - nm) + __expf(s - nm);
    m = nm;
  }
  float inv = 1.f / Z;
  for (int r = 0; r < SS; ++r) {
    float s = (float)p[(size_t)r * SS];
    p[(size_t)r * SS] = (_Float16)(__expf(s - m) * inv);
  }
}

// ---------------- WMMA GEMM ----------------
// C[M,N] = A[M,K] (f16, row-major, lda) @ Bt[N,K]^T (f16, row-major, ldb) + bias
// OUT_MODE 0: f16 row-major(ldc)   1: f32 row-major(ldc)
//          2: f16 tile-transposed scatter for V^T, via LDS bounce:
//             (row=b*1024+s, col=h*128+d) -> [((b*16+h)*128+d)*1024 + s]
template <int OUT_MODE>
__global__ __launch_bounds__(256) void gemm_wmma_f16(
    const _Float16* __restrict__ A, const _Float16* __restrict__ Bt,
    const float* __restrict__ bias, void* __restrict__ Cout,
    int M, int N, int K, int lda, int ldb, int ldc,
    long sA, long sB, long sC) {
  // 24KB: double-buffered A (2x8KB) + B (2x4KB); reused as 16KB transpose tile.
  __shared__ __align__(16) char smem[24576];

  const int tid  = threadIdx.x;
  const int wid  = tid >> 5;
  const int lane = tid & 31;
  const int half = lane >> 4;
  const int lm   = lane & 15;

  const int bm = blockIdx.y * 128;
  const int bn = blockIdx.x * 64;
  const long z = blockIdx.z;
  A  += z * sA;
  Bt += z * sB;

  const int waveM = (wid >> 1) * 32;
  const int waveN = (wid & 1) * 32;

  v8f acc[2][2];
#pragma unroll
  for (int i = 0; i < 2; ++i)
#pragma unroll
    for (int j = 0; j < 2; ++j)
#pragma unroll
      for (int r = 0; r < 8; ++r) acc[i][j][r] = 0.f;

  const int r0  = tid >> 2;           // 0..63
  const int r1  = r0 + 64;            // 64..127 (A only)
  const int off = (tid & 3) * 8;      // f16 elements, 16B chunks

  auto lsA = [&](int buf) { return (_Float16*)(smem + buf * 8192); };
  auto lsB = [&](int buf) { return (_Float16*)(smem + 16384 + buf * 4096); };

  auto stage = [&](int buf, int k0) {
    const _Float16* ga0 = A + (long)(bm + r0) * lda + k0 + off;
    const _Float16* ga1 = A + (long)(bm + r1) * lda + k0 + off;
    const _Float16* gb0 = Bt + (long)(bn + r0) * ldb + k0 + off;
    _Float16* la0 = lsA(buf) + r0 * 32 + off;
    _Float16* la1 = lsA(buf) + r1 * 32 + off;
    _Float16* lb0 = lsB(buf) + r0 * 32 + off;
#if USE_ASYNC_LDS
    __builtin_amdgcn_global_load_async_to_lds_b128((AS1 v4i*)ga0, (AS3 v4i*)la0, 0, 0);
    __builtin_amdgcn_global_load_async_to_lds_b128((AS1 v4i*)ga1, (AS3 v4i*)la1, 0, 0);
    __builtin_amdgcn_global_load_async_to_lds_b128((AS1 v4i*)gb0, (AS3 v4i*)lb0, 0, 0);
#else
    uint4 a0 = *(const uint4*)ga0;
    uint4 a1 = *(const uint4*)ga1;
    uint4 b0 = *(const uint4*)gb0;
    *(uint4*)la0 = a0;
    *(uint4*)la1 = a1;
    *(uint4*)lb0 = b0;
#endif
  };

  const int nk = K >> 5;
  stage(0, 0);
#if USE_ASYNC_LDS
  wait_async0();
#endif
  __syncthreads();

  for (int it = 0; it < nk; ++it) {
    const int buf = it & 1;
    if (it + 1 < nk) stage(buf ^ 1, (it + 1) * 32);
    if (it + 2 < nk) {  // warm L2 two tiles ahead (global_prefetch_b8)
      __builtin_prefetch(A + (long)(bm + r0) * lda + (it + 2) * 32 + off, 0, 1);
      __builtin_prefetch(Bt + (long)(bn + r0) * ldb + (it + 2) * 32 + off, 0, 1);
    }

    // Fragment layouts per CDNA5 ISA 7.12.2 (16-bit A 16x32 / B 32x16)
    Frag af[2], bf[2];
#pragma unroll
    for (int i = 0; i < 2; ++i) {
      const _Float16* pa = lsA(buf) + (waveM + i * 16 + lm) * 32 + half * 8;
      af[i].q[0] = *(const uint4*)pa;          // K = 8*half + 0..7
      af[i].q[1] = *(const uint4*)(pa + 16);   // K = 16 + 8*half + 0..7
      const _Float16* pb = lsB(buf) + (waveN + i * 16 + lm) * 32 + half * 8;
      bf[i].q[0] = *(const uint4*)pb;          // K = 8*half + 0..7
      bf[i].q[1] = *(const uint4*)(pb + 16);   // K = 16 + 8*half + 0..7
    }
#pragma unroll
    for (int i = 0; i < 2; ++i)
#pragma unroll
      for (int j = 0; j < 2; ++j)
        acc[i][j] = __builtin_amdgcn_wmma_f32_16x16x32_f16(
            false, af[i].h, false, bf[j].h, (short)0, acc[i][j], false, false);

    if (it + 1 < nk) {
#if USE_ASYNC_LDS
      wait_async0();
#endif
      __syncthreads();
    }
  }

  if (OUT_MODE == 2) {
    // Bounce the 128x64 tile through LDS, store transposed with coalesced rows.
    __syncthreads();  // everyone done reading staging buffers
    _Float16* lsT = (_Float16*)smem;  // [64 cols][128 rows] = 16KB
#pragma unroll
    for (int i = 0; i < 2; ++i)
#pragma unroll
      for (int j = 0; j < 2; ++j) {
        const int dl = waveN + j * 16 + lm;            // local col 0..63
        const float bv = bias ? bias[bn + dl] : 0.f;
#pragma unroll
        for (int r = 0; r < 8; ++r) {
          const int sl = waveM + i * 16 + r + 8 * half;  // local row 0..127
          lsT[dl * 128 + sl] = (_Float16)(acc[i][j][r] + bv);
        }
      }
    __syncthreads();
    const int b = bm >> 10, s0 = bm & 1023;
    const int h = bn >> 7, d0 = bn & 127;
    const int dl = tid >> 2;            // 0..63
    const int so = (tid & 3) * 32;      // 32 f16 = 64B per thread
    _Float16* dst = (_Float16*)Cout +
                    (((long)((b * HH + h) * HDD + d0 + dl)) << 10) + s0 + so;
    const uint4* src = (const uint4*)(lsT + dl * 128 + so);
#pragma unroll
    for (int q = 0; q < 4; ++q) ((uint4*)dst)[q] = src[q];
    return;
  }

#pragma unroll
  for (int i = 0; i < 2; ++i)
#pragma unroll
    for (int j = 0; j < 2; ++j) {
      const int col = bn + waveN + j * 16 + lm;
      const float bv = bias ? bias[col] : 0.f;
#pragma unroll
      for (int r = 0; r < 8; ++r) {
        const int row = bm + waveM + i * 16 + r + 8 * half;
        const float v = acc[i][j][r] + bv;
        if (OUT_MODE == 0) {
          ((_Float16*)Cout)[z * sC + (long)row * ldc + col] = (_Float16)v;
        } else {
          ((float*)Cout)[z * sC + (long)row * ldc + col] = v;
        }
      }
    }
}

// ---------------- launch ----------------
extern "C" void kernel_launch(void* const* d_in, const int* in_sizes, int n_in,
                              void* d_out, int out_size, void* d_ws, size_t ws_size,
                              hipStream_t stream) {
  const float* query = (const float*)d_in[0];
  const float* key   = (const float*)d_in[1];
  const float* value = (const float*)d_in[2];
  const float* W_A = (const float*)d_in[3];
  const float* W_B = (const float*)d_in[4];
  const float* W_C = (const float*)d_in[5];
  const float* W_D = (const float*)d_in[6];
  const float* W_E = (const float*)d_in[7];
  const float* W_F = (const float*)d_in[8];
  const float* Wq = (const float*)d_in[9];  const float* bq = (const float*)d_in[10];
  const float* Wk = (const float*)d_in[11]; const float* bk = (const float*)d_in[12];
  const float* Wv = (const float*)d_in[13]; const float* bv = (const float*)d_in[14];
  const float* Wo = (const float*)d_in[15]; const float* bo = (const float*)d_in[16];
  float* out = (float*)d_out;

  char* ws = (char*)d_ws;
  size_t o = 0;
  auto alloc = [&](size_t bytes) { size_t r = o; o += (bytes + 255) & ~(size_t)255; return r; };
  _Float16* xq16 = (_Float16*)(ws + alloc((size_t)MTOT * DD * 2));
  _Float16* xk16 = (_Float16*)(ws + alloc((size_t)MTOT * DD * 2));
  _Float16* xv16 = (_Float16*)(ws + alloc((size_t)MTOT * DD * 2));
  _Float16* wqT  = (_Float16*)(ws + alloc((size_t)DD * PP * 2));
  _Float16* wkT  = (_Float16*)(ws + alloc((size_t)DD * PP * 2));
  _Float16* wvT  = (_Float16*)(ws + alloc((size_t)DD * PP * 2));
  _Float16* woT  = (_Float16*)(ws + alloc((size_t)PP * DD * 2));
  _Float16* qp   = (_Float16*)(ws + alloc((size_t)MTOT * PP * 2));
  _Float16* kp   = (_Float16*)(ws + alloc((size_t)MTOT * PP * 2));
  _Float16* vpT  = (_Float16*)(ws + alloc((size_t)MTOT * PP * 2));
  _Float16* scr  = (_Float16*)(ws + alloc((size_t)BB * HH * SS * SS * 2));
  _Float16* xatt = (_Float16*)(ws + alloc((size_t)MTOT * PP * 2));

  // 1) f32 -> f16 activations
  {
    int n4 = MTOT * DD / 4;
    dim3 g((n4 + 255) / 256), b(256);
    cvt_f32_to_f16_x4<<<g, b, 0, stream>>>((const float4*)query, xq16, n4);
    cvt_f32_to_f16_x4<<<g, b, 0, stream>>>((const float4*)key,   xk16, n4);
    cvt_f32_to_f16_x4<<<g, b, 0, stream>>>((const float4*)value, xv16, n4);
  }
  // 2) fold LoRA into weights, store transposed f16
  {
    dim3 b(256);
    dim3 g1(((size_t)DD * PP + 255) / 256);
    fuse_weight_T<<<g1, b, 0, stream>>>(Wq, W_A, W_B, wqT, DD, PP);
    fuse_weight_T<<<g1, b, 0, stream>>>(Wk, W_C, W_D, wkT, DD, PP);
    fuse_weight_T<<<g1, b, 0, stream>>>(Wv, W_E, W_F, wvT, DD, PP);
    dim3 g2(((size_t)PP * DD + 255) / 256);
    fuse_weight_T<<<g2, b, 0, stream>>>(Wo, nullptr, nullptr, woT, PP, DD);
  }
  // 3) projections: qp/kp row-major (4096x2048); v written directly transposed per head
  {
    dim3 b(256), g(PP / 64, MTOT / 128, 1);
    gemm_wmma_f16<0><<<g, b, 0, stream>>>(xq16, wqT, bq, qp, MTOT, PP, DD, DD, DD, PP, 0, 0, 0);
    gemm_wmma_f16<0><<<g, b, 0, stream>>>(xk16, wkT, bk, kp, MTOT, PP, DD, DD, DD, PP, 0, 0, 0);
    gemm_wmma_f16<2><<<g, b, 0, stream>>>(xv16, wvT, bv, vpT, MTOT, PP, DD, DD, DD, 0, 0, 0, 0);
  }
  // 4) scores[z] = Qh @ Kh^T  (z = b*H + h; head slabs are contiguous S x HD)
  {
    dim3 b(256), g(SS / 64, SS / 128, BB * HH);
    gemm_wmma_f16<0><<<g, b, 0, stream>>>(qp, kp, nullptr, scr, SS, SS, HDD, HDD, HDD, SS,
                                          (long)SS * HDD, (long)SS * HDD, (long)SS * SS);
  }
  // 5) softmax over the query axis (columns), in place
  {
    dim3 b(256), g(SS / 256, BB * HH);
    col_softmax<<<g, b, 0, stream>>>(scr);
  }
  // 6) x[z] = P @ Vh  (Vh given transposed: vpT is HD x S per head)
  {
    dim3 b(256), g(HDD / 64, SS / 128, BB * HH);
    gemm_wmma_f16<0><<<g, b, 0, stream>>>(scr, vpT, nullptr, xatt, SS, HDD, SS, SS, SS, HDD,
                                          (long)SS * SS, (long)SS * HDD, (long)SS * HDD);
  }
  // 7) out = X @ Wo + bo  (f32 output)
  {
    dim3 b(256), g(DD / 64, MTOT / 128, 1);
    gemm_wmma_f16<1><<<g, b, 0, stream>>>(xatt, woT, bo, out, MTOT, DD, PP, PP, PP, DD, 0, 0, 0);
  }
}
